// ScaledDotProduct_51659866636826
// MI455X (gfx1250) — matrix operand order repaired and checked by reading, not compiled, not run on hardware
//
#include <hip/hip_runtime.h>

typedef _Float16 v16h __attribute__((ext_vector_type(16)));
typedef _Float16 v8h  __attribute__((ext_vector_type(8)));
typedef _Float16 v2h  __attribute__((ext_vector_type(2)));
typedef float    v8f  __attribute__((ext_vector_type(8)));
typedef v16h v16h_a __attribute__((aligned(16)));   // 16B-aligned 32B LDS accesses

#define HEADS  16
#define SEQ    4096
#define DHEAD  64
#define BM     128     // query rows per workgroup (4 waves x 32 rows)
#define BN     32      // keys per kv tile
#define NWAVE  4
#define KS_PAD 72      // 64 + 8 halves per K row in LDS
#define VT_PAD 40      // 32 + 8 halves per Vt row in LDS
#define P_PAD  40      // 32 + 8 halves per P row in LDS
#define NEGBIG (-1.0e30f)

// xor-shuffle within each 16-lane row via v_permlane16_b32 (no DS traffic)
__device__ __forceinline__ float plane16(float v, unsigned lo, unsigned hi) {
  unsigned u = __float_as_uint(v);
  u = __builtin_amdgcn_permlane16(u, u, lo, hi, false, false);
  return __uint_as_float(u);
}
__device__ __forceinline__ float rowmax16(float v) {
  v = fmaxf(v, plane16(v, 0x67452301u, 0xEFCDAB89u));  // xor 1
  v = fmaxf(v, plane16(v, 0x54761032u, 0xDCFE98BAu));  // xor 2
  v = fmaxf(v, plane16(v, 0x32107654u, 0xBA98FEDCu));  // xor 4
  v = fmaxf(v, plane16(v, 0xFEDCBA98u, 0x76543210u));  // xor 8
  return v;
}

__launch_bounds__(128, 1)
__global__ void fa_fwd_causal(const float* __restrict__ Q,
                              const float* __restrict__ K,
                              const float* __restrict__ V,
                              float* __restrict__ O)
{
  __shared__ _Float16 Ks[2][BN][KS_PAD];        // K tile, row-major [key][d]
  __shared__ _Float16 Vt[2][DHEAD][VT_PAD];     // V tile transposed [d][key]
  __shared__ _Float16 Ps[NWAVE][32][P_PAD];     // per-wave P tiles [m][key]

  const int tid  = threadIdx.x;
  const int lane = tid & 31;
  const int wave = tid >> 5;          // 0..3
  const int g    = lane >> 4;         // half-wave group (0/1)
  const int n    = lane & 15;
  const int head = blockIdx.y;
  const int qbase = blockIdx.x * BM;
  const int qb    = qbase + wave * 32;           // wave's first query row

  const size_t hoff = (size_t)head * SEQ * DHEAD;

  // staging coords: each thread owns 2 adjacent key rows x 8 d-columns
  const int sr0 = (tid >> 3) * 2;       // 0,2,...,30
  const int sc  = (tid & 7) * 8;        // 0..56

  // log2-domain scores: fold 1/sqrt(d) * log2(e) into Q
  const float scale = 0.125f * 1.44269504088896340736f;

  // ---- Q fragments (A layout) for two 16-row sets ----
  v16h a[2][2];
#pragma unroll
  for (int s = 0; s < 2; ++s) {
    const float* qrow = Q + hoff + (size_t)(qb + 16 * s + n) * DHEAD;
#pragma unroll
    for (int h = 0; h < 16; ++h) {
      const int d0 = h + 16 * g;
      a[s][0][h] = (_Float16)(qrow[d0]      * scale);
      a[s][1][h] = (_Float16)(qrow[d0 + 32] * scale);
    }
  }

  // all-ones B fragment: osum = P x 1 replicates row-sum into every lane
  v16h onesb;
#pragma unroll
  for (int h = 0; h < 16; ++h) onesb[h] = (_Float16)1.0f;

  // ---- softmax state + output accumulators (2 row sets) ----
  float m_i[2][8];
  v8f o[2][4], osum[2];
#pragma unroll
  for (int s = 0; s < 2; ++s) {
#pragma unroll
    for (int r = 0; r < 8; ++r) { m_i[s][r] = NEGBIG; osum[s][r] = 0.0f; }
#pragma unroll
    for (int t = 0; t < 4; ++t)
#pragma unroll
      for (int r = 0; r < 8; ++r) o[s][t][r] = 0.0f;
  }

  const int ntiles = (qbase + BM) / BN;   // causal bound for this block

  // register staging of one KV tile (2 rows x 8 cols of K and V each)
  float rka[8], rkb[8], rva[8], rvb[8];
  auto stage_load = [&](int kb) {
    const float* kp0 = K + hoff + (size_t)(kb + sr0) * DHEAD + sc;
    const float* kp1 = kp0 + DHEAD;
    const float* vp0 = V + hoff + (size_t)(kb + sr0) * DHEAD + sc;
    const float* vp1 = vp0 + DHEAD;
#pragma unroll
    for (int i = 0; i < 8; ++i) {
      rka[i] = kp0[i]; rkb[i] = kp1[i];
      rva[i] = vp0[i]; rvb[i] = vp1[i];
    }
  };
  auto stage_store = [&](int buf) {
    v8h ha, hb;
#pragma unroll
    for (int i = 0; i < 8; ++i) {
      ha[i] = (_Float16)rka[i];
      hb[i] = (_Float16)rkb[i];
    }
    *(v8h*)&Ks[buf][sr0][sc]     = ha;
    *(v8h*)&Ks[buf][sr0 + 1][sc] = hb;
    // V transposed: two adjacent key columns -> one packed b32 store per d
#pragma unroll
    for (int j = 0; j < 8; ++j) {
      v2h p;
      p[0] = (_Float16)rva[j];
      p[1] = (_Float16)rvb[j];
      *(v2h*)&Vt[buf][sc + j][sr0] = p;
    }
  };

  // prologue: stage tile 0 into buffer 0
  stage_load(0);
  stage_store(0);

  for (int kt = 0; kt < ntiles; ++kt) {
    const int kb  = kt * BN;
    const int cur = kt & 1;

    // prefetch next tile into registers (latency hidden behind barrier+compute)
    if (kt + 1 < ntiles) stage_load(kb + BN);

    __syncthreads();   // buf[cur] stores from previous iteration now visible

#pragma unroll
    for (int s = 0; s < 2; ++s) {
      const int qbs   = qb + 16 * s;       // this set's first row
      const int qrow0 = qbs + 8 * g;       // row for accumulator idx r

      // ---- scores S = Q K^T (log2 domain): 2 subtiles x 2 k-chunks ----
      v8f c0, c1;
#pragma unroll
      for (int r = 0; r < 8; ++r) { c0[r] = 0.0f; c1[r] = 0.0f; }
#pragma unroll
      for (int kk = 0; kk < 2; ++kk) {
        const int off = 32 * kk + 16 * g;
        v16h b0 = *(const v16h_a*)&Ks[cur][n][off];
        v16h b1 = *(const v16h_a*)&Ks[cur][16 + n][off];
        c0 = __builtin_amdgcn_wmma_f32_16x16x32_f16(false, a[s][kk], false, b0,
                                                    (short)0, c0, false, false);
        c1 = __builtin_amdgcn_wmma_f32_16x16x32_f16(false, a[s][kk], false, b1,
                                                    (short)0, c1, false, false);
      }

      // ---- causal mask (uniform per wave/set whether needed) ----
      if (kb + BN - 1 > qbs) {
#pragma unroll
        for (int r = 0; r < 8; ++r) {
          const int q = qrow0 + r;
          if (kb + n      > q) c0[r] = NEGBIG;
          if (kb + 16 + n > q) c1[r] = NEGBIG;
        }
      }

      // ---- online softmax: max via permlane16; exp2 (scores already log2) ----
      float alpha[8];
#pragma unroll
      for (int r = 0; r < 8; ++r) {
        const float f01  = c0[r] > c1[r] ? c0[r] : c1[r];  // wmma outputs: no sNaN
        const float mx   = rowmax16(f01);
        const float mnew = fmaxf(m_i[s][r], mx);
        alpha[r] = __builtin_amdgcn_exp2f(m_i[s][r] - mnew);
        m_i[s][r] = mnew;
        c0[r] = __builtin_amdgcn_exp2f(c0[r] - mnew);
        c1[r] = __builtin_amdgcn_exp2f(c1[r] - mnew);
      }
#pragma unroll
      for (int r = 0; r < 8; ++r) {
        osum[s][r] *= alpha[r];
#pragma unroll
        for (int t = 0; t < 4; ++t) o[s][t][r] *= alpha[r];
      }

      // ---- P: C-layout -> A-layout via LDS, columns in key order ----
      {
        const int pos0 = n;
#pragma unroll
        for (int r = 0; r < 8; ++r) {
          Ps[wave][16 * s + r + 8 * g][pos0]     = (_Float16)c0[r];   // k = n
          Ps[wave][16 * s + r + 8 * g][pos0 + 16] = (_Float16)c1[r];   // k = 16+n
        }
      }
      v16h pa = *(const v16h_a*)&Ps[wave][16 * s + n][16 * g];

      // ---- O += P*V (4 d-subtiles) and row-sum accumulator ----
      osum[s] = __builtin_amdgcn_wmma_f32_16x16x32_f16(false, pa, false, onesb,
                                                       (short)0, osum[s], false, false);
#pragma unroll
      for (int t = 0; t < 4; ++t) {
        v16h vb = *(const v16h_a*)&Vt[cur][16 * t + n][16 * g];
        o[s][t] = __builtin_amdgcn_wmma_f32_16x16x32_f16(false, pa, false, vb,
                                                         (short)0, o[s][t], false, false);
      }
    }

    // write next tile into the other buffer (other waves still read buf[cur])
    if (kt + 1 < ntiles) stage_store(cur ^ 1);
  }

  // ---- normalize and store ----
#pragma unroll
  for (int s = 0; s < 2; ++s) {
    const int qrow0 = qb + 16 * s + 8 * g;
    float inv[8];
#pragma unroll
    for (int r = 0; r < 8; ++r) inv[r] = 1.0f / osum[s][r];
#pragma unroll
    for (int t = 0; t < 4; ++t)
#pragma unroll
      for (int r = 0; r < 8; ++r)
        O[hoff + (size_t)(qrow0 + r) * DHEAD + 16 * t + n] = o[s][t][r] * inv[r];
  }
}

extern "C" void kernel_launch(void* const* d_in, const int* in_sizes, int n_in,
                              void* d_out, int out_size, void* d_ws, size_t ws_size,
                              hipStream_t stream) {
  (void)in_sizes; (void)n_in; (void)out_size; (void)d_ws; (void)ws_size;
  const float* q = (const float*)d_in[0];
  const float* k = (const float*)d_in[1];
  const float* v = (const float*)d_in[2];
  float* out = (float*)d_out;
  dim3 grid(SEQ / BM, HEADS);
  fa_fwd_causal<<<grid, dim3(128), 0, stream>>>(q, k, v, out);
}
